// Net_GCN_ASAP_GlobalPooling_DO_56727928046283
// MI455X (gfx1250) — compile-verified, matrix operand order checked by reading, not run on hardware
//
#include <hip/hip_runtime.h>
#include <hip/hip_bf16.h>

// =====================================================================
// GCN + ASAP pooling forward for MI455X (gfx1250), wave32 + WMMA.
//
// All dense matmuls run through v_wmma_f32_16x16x32_bf16 with BOTH
// operands stored K-major (B pre-transposed), so each lane feeds the
// matrix pipe with unpredicated global_load_b128s:
//   lane elements e0..7  = 8 contiguous bf16 at [row, k0 + half*8]
//   lane elements e8..15 = 8 contiguous bf16 at [row, k0 + 16 + half*8]
// Each wave computes a 32x32 output tile (2x2 WMMA register blocking):
// per 32-wide K step, 8 b128 loads feed 4 WMMAs (2x operand reuse vs a
// single 16x16 tile), which matters because the big NxN operands are
// L2-resident and the kernel is otherwise L2-bandwidth bound.
// K is zero-padded to a multiple of 32 in the bf16 convert kernels, M is
// always a multiple of 16; row/col subtile clamping is uniform per wave
// so EXEC is all-ones at every WMMA.
// Anorm and A+I are symmetric (ring lattice; S^T M S preserves symmetry),
// so they serve as their own transposes on the B side.
//
// Irregular graph ops (segment max/softmax, LEConv fitness, per-graph
// top-k, binarize) are fp32 VALU kernels over the dense adjacency
// (64 MB fp32 at stage 1 -- L2-resident on the 192 MB L2).
//
// Param indexing assumes dict-insertion-order flattening of
// setup_inputs(): x, edge_index, batch, params(b1_gnn, b1_asap, b2_gnn,
// b2_asap, b3_gnn, b3_asap, b4_gnn, fc).
// =====================================================================

typedef __attribute__((ext_vector_type(16))) __bf16 v16bf;
typedef __attribute__((ext_vector_type(8)))  __bf16 v8bf;
typedef __attribute__((ext_vector_type(8)))  float  v8f;

__device__ inline __bf16 f2bf(float f) {
  union { float f; unsigned u; } c; c.f = f;
  unsigned u = c.u;
  unsigned r = u + 0x7FFFu + ((u >> 16) & 1u);   // round-to-nearest-even
  unsigned short h = (unsigned short)(r >> 16);
  __bf16 out; __builtin_memcpy(&out, &h, 2);
  return out;
}

// two 16-byte loads -> one 16-element WMMA operand fragment
__device__ inline v16bf ld_frag(const __bf16* __restrict__ p) {
  v8bf lo = *(const v8bf*)(p);
  v8bf hi = *(const v8bf*)(p + 16);
  return __builtin_shufflevector(lo, hi, 0,1,2,3,4,5,6,7,8,9,10,11,12,13,14,15);
}

// ---------------------------------------------------------------------
// GEMM: C[MxN] = A[MxKp](bf16,K-major) * Bt[NxKp](bf16,K-major)^T
// flags: 1=add bias[n], 2=relu, 8=emit bf16 to Ch (else f32 to Cf).
// One wave per 32x32 tile (2x2 blocking), 4 waves per block along M.
// Requires: Kp % 32 == 0, M % 16 == 0, Bt padded to ceil(N/16)*16 rows,
// lda/ldb multiples of 8 (16-byte aligned rows).
// ---------------------------------------------------------------------
__global__ __launch_bounds__(128) void k_gemm(
    const __bf16* __restrict__ Am, const __bf16* __restrict__ Bt,
    const float* __restrict__ bias, float* __restrict__ Cf,
    __bf16* __restrict__ Ch, int M, int N, int Kp,
    int lda, int ldb, int ldc, int flags)
{
  const int lane = threadIdx.x;                    // 0..31
  const int half = lane >> 4;
  const int mn   = lane & 15;
  const int tm   = (blockIdx.y * 4 + threadIdx.y) << 5;
  const int tn   = blockIdx.x << 5;
  if (tm >= M) return;                             // uniform per wave

  const int Np = (N + 15) & ~15;                   // padded B rows
  const int m1 = (tm + 16 < M)  ? tm + 16 : tm;    // clamped row subtile
  const int n1 = (tn + 16 < Np) ? tn + 16 : tn;    // clamped col subtile

  const __bf16* ap0 = Am + (size_t)(tm + mn) * lda + (half << 3);
  const __bf16* ap1 = Am + (size_t)(m1 + mn) * lda + (half << 3);
  const __bf16* bp0 = Bt + (size_t)(tn + mn) * ldb + (half << 3);
  const __bf16* bp1 = Bt + (size_t)(n1 + mn) * ldb + (half << 3);

  v8f acc00 = {}, acc01 = {}, acc10 = {}, acc11 = {};
#pragma unroll 2
  for (int k0 = 0; k0 < Kp; k0 += 32) {
    v16bf a0 = ld_frag(ap0 + k0);
    v16bf a1 = ld_frag(ap1 + k0);
    v16bf b0 = ld_frag(bp0 + k0);
    v16bf b1 = ld_frag(bp1 + k0);
    acc00 = __builtin_amdgcn_wmma_f32_16x16x32_bf16(false, a0, false, b0, (short)0, acc00, false, false);
    acc01 = __builtin_amdgcn_wmma_f32_16x16x32_bf16(false, a0, false, b1, (short)0, acc01, false, false);
    acc10 = __builtin_amdgcn_wmma_f32_16x16x32_bf16(false, a1, false, b0, (short)0, acc10, false, false);
    acc11 = __builtin_amdgcn_wmma_f32_16x16x32_bf16(false, a1, false, b1, (short)0, acc11, false, false);
  }

  const bool row1ok = (tm + 16 < M);
#pragma unroll
  for (int r = 0; r < 8; ++r) {
    const int mo0 = tm + r + (half << 3);
    const int mo1 = tm + 16 + r + (half << 3);
    const int no0 = tn + mn;
    const int no1 = tn + 16 + mn;
    float v;
    if (no0 < N) {
      v = acc00[r];
      if (flags & 1) v += bias[no0];
      if (flags & 2) v = v > 0.f ? v : 0.f;
      if (flags & 8) Ch[(size_t)mo0 * ldc + no0] = f2bf(v);
      else           Cf[(size_t)mo0 * ldc + no0] = v;
      if (row1ok) {
        v = acc10[r];
        if (flags & 1) v += bias[no0];
        if (flags & 2) v = v > 0.f ? v : 0.f;
        if (flags & 8) Ch[(size_t)mo1 * ldc + no0] = f2bf(v);
        else           Cf[(size_t)mo1 * ldc + no0] = v;
      }
    }
    if (no1 < N) {
      v = acc01[r];
      if (flags & 1) v += bias[no1];
      if (flags & 2) v = v > 0.f ? v : 0.f;
      if (flags & 8) Ch[(size_t)mo0 * ldc + no1] = f2bf(v);
      else           Cf[(size_t)mo0 * ldc + no1] = v;
      if (row1ok) {
        v = acc11[r];
        if (flags & 1) v += bias[no1];
        if (flags & 2) v = v > 0.f ? v : 0.f;
        if (flags & 8) Ch[(size_t)mo1 * ldc + no1] = f2bf(v);
        else           Cf[(size_t)mo1 * ldc + no1] = v;
      }
    }
  }
}

// ------------------------- converts / setup --------------------------
// dst[i*ldd + k] = k<C ? bf16(src[i*C+k]) : 0     (row-padded copy)
__global__ void k_conv_pad(const float* __restrict__ src, __bf16* __restrict__ dst,
                           int C, int ldd) {
  const int i = blockIdx.y;
  const int k = blockIdx.x * 256 + threadIdx.x;
  if (k < ldd) dst[(size_t)i * ldd + k] = f2bf(k < C ? src[(size_t)i * C + k] : 0.f);
}

// Transposed+padded: src R x C row-major -> dst (Cp rows) x ldd bf16,
// dst[c*ldd + r] = (r<R && c<C) ? bf16(src[r*C+c]) : 0
__global__ void k_convT(const float* __restrict__ src, __bf16* __restrict__ dst,
                        int R, int C, int ldd) {
  const int c = blockIdx.y;                 // 0..Cp-1
  const int r = blockIdx.x * 256 + threadIdx.x;
  if (r < ldd) {
    float v = (r < R && c < C) ? src[(size_t)r * C + c] : 0.f;
    dst[(size_t)c * ldd + r] = f2bf(v);
  }
}

__global__ void k_zero_f32(float* __restrict__ p, int n) {
  int i = blockIdx.x * 256 + threadIdx.x;
  if (i < n) p[i] = 0.f;
}

__global__ void k_build_A0(const int* __restrict__ ei, int E, float* __restrict__ A, int N) {
  int e = blockIdx.x * 256 + threadIdx.x;
  if (e < E) A[(size_t)ei[e] * N + ei[E + e]] = 1.f;
}

// di[i] = rsqrt(1 + sum_j A[i,j])   (Ah = A + I)
__global__ void k_degree(const float* __restrict__ A, float* __restrict__ di, int N) {
  const int i = blockIdx.x, t = threadIdx.x;
  __shared__ float rs[256];
  float s = 0.f;
  for (int j = t; j < N; j += 256) s += A[(size_t)i * N + j];
  rs[t] = s; __syncthreads();
  for (int sh = 128; sh > 0; sh >>= 1) { if (t < sh) rs[t] += rs[t + sh]; __syncthreads(); }
  if (t == 0) di[i] = rsqrtf(rs[0] + 1.f);
}

// Ah[i,j] = bf16( di[i]*di[j]*(A+I)[i,j] )   (symmetric)
__global__ void k_anorm(const float* __restrict__ A, const float* __restrict__ di,
                        __bf16* __restrict__ Ah, int N) {
  const int idx = blockIdx.x * 256 + threadIdx.x;
  if (idx >= N * N) return;
  const int i = idx / N, j = idx - i * N;
  float v = (A[idx] != 0.f || i == j) ? di[i] * di[j] : 0.f;
  Ah[idx] = f2bf(v);
}

// Asl[i,j] = bf16( (A+I)[i,j] )   (weights all 1, symmetric)
__global__ void k_asl(const float* __restrict__ A, __bf16* __restrict__ Ah, int N) {
  const int idx = blockIdx.x * 256 + threadIdx.x;
  if (idx >= N * N) return;
  const int i = idx / N, j = idx - i * N;
  Ah[idx] = f2bf((A[idx] != 0.f || i == j) ? 1.f : 0.f);
}

// A[p,q] = (Anew[p,q] > 0 && p != q) ? 1 : 0
__global__ void k_binarize(const float* __restrict__ An, float* __restrict__ A, int K) {
  const int idx = blockIdx.x * 256 + threadIdx.x;
  if (idx >= K * K) return;
  const int p = idx / K, q = idx - p * K;
  A[idx] = (An[idx] > 0.f && p != q) ? 1.f : 0.f;
}

// ---------------------- BatchNorm(relu(Z)) ---------------------------
__global__ void k_bn(const float* __restrict__ Z, const float* __restrict__ g,
                     const float* __restrict__ be, float* __restrict__ Y,
                     int N, int d) {
  const int f = blockIdx.x, t = threadIdx.x;
  __shared__ float rs[256], rq[256];
  float s = 0.f, q = 0.f;
  for (int i = t; i < N; i += 256) {
    float v = fmaxf(Z[(size_t)i * d + f], 0.f);
    s += v; q += v * v;
  }
  rs[t] = s; rq[t] = q; __syncthreads();
  for (int sh = 128; sh > 0; sh >>= 1) {
    if (t < sh) { rs[t] += rs[t + sh]; rq[t] += rq[t + sh]; }
    __syncthreads();
  }
  __shared__ float mean, inv;
  if (t == 0) {
    float m = rs[0] / N;
    float var = rq[0] / N - m * m;
    mean = m; inv = rsqrtf(var + 1e-5f);
  }
  __syncthreads();
  const float gg = g[f], bb = be[f];
  for (int i = t; i < N; i += 256) {
    float v = fmaxf(Z[(size_t)i * d + f], 0.f);
    Y[(size_t)i * d + f] = (v - mean) * inv * gg + bb;
  }
}

// y[i] = sum_f X[i,f]*w[f] (+b[0])
__global__ void k_rowdot(const float* __restrict__ Xm, const float* __restrict__ w,
                         const float* __restrict__ b, float* __restrict__ y, int d) {
  const int i = blockIdx.x, t = threadIdx.x;
  __shared__ float rs[256];
  float s = 0.f;
  for (int f = t; f < d; f += 256) s += Xm[(size_t)i * d + f] * w[f];
  rs[t] = s; __syncthreads();
  for (int sh = 128; sh > 0; sh >>= 1) { if (t < sh) rs[t] += rs[t + sh]; __syncthreads(); }
  if (t == 0) y[i] = rs[0] + (b ? b[0] : 0.f);
}

// x_q[i,f] = max over in-neighbors j (A symmetric: row scan) of X[j,f]
__global__ void k_segmax(const float* __restrict__ A, const float* __restrict__ X,
                         float* __restrict__ XQ, int N, int d) {
  const int i = blockIdx.x, f = threadIdx.x;   // blockDim == d
  __shared__ float acol[128];
  float mx = -1e30f; int any = 0;
  for (int j0 = 0; j0 < N; j0 += 128) {
    if (f < 128 && (j0 + f) < N) acol[f] = A[(size_t)i * N + j0 + f];
    __syncthreads();
    const int lim = min(128, N - j0);
    for (int jo = 0; jo < lim; ++jo) {
      const int j = j0 + jo;
      if (acol[jo] != 0.f || j == i) { mx = fmaxf(mx, X[(size_t)j * d + f]); any = 1; }
    }
    __syncthreads();
  }
  XQ[(size_t)i * d + f] = any ? mx : 0.f;
}

// Segment softmax over in-edges of i: St[i,j] = softmax_j(lrelu(u[i]+w[j]))
__global__ void k_scores(const float* __restrict__ A, const float* __restrict__ u,
                         const float* __restrict__ w, __bf16* __restrict__ St, int N) {
  const int i = blockIdx.x, t = threadIdx.x;
  __shared__ float red[256];
  const float ui = u[i];
  float mx = -1e30f;
  for (int j = t; j < N; j += 256)
    if (A[(size_t)i * N + j] != 0.f || j == i) {
      float s = ui + w[j]; s = (s > 0.f) ? s : 0.2f * s;
      mx = fmaxf(mx, s);
    }
  red[t] = mx; __syncthreads();
  for (int sh = 128; sh > 0; sh >>= 1) { if (t < sh) red[t] = fmaxf(red[t], red[t + sh]); __syncthreads(); }
  const float smax = red[0]; __syncthreads();
  float sum = 0.f;
  for (int j = t; j < N; j += 256)
    if (A[(size_t)i * N + j] != 0.f || j == i) {
      float s = ui + w[j]; s = (s > 0.f) ? s : 0.2f * s;
      sum += __expf(s - smax);
    }
  red[t] = sum; __syncthreads();
  for (int sh = 128; sh > 0; sh >>= 1) { if (t < sh) red[t] += red[t + sh]; __syncthreads(); }
  const float den = red[0]; __syncthreads();
  for (int j = t; j < N; j += 256) {
    float sc = 0.f;
    if (A[(size_t)i * N + j] != 0.f || j == i) {
      float s = ui + w[j]; s = (s > 0.f) ? s : 0.2f * s;
      sc = __expf(s - smax) / den;
    }
    St[(size_t)i * N + j] = f2bf(sc);
  }
}

// LEConv fitness: fit[i] = sigmoid( sum_{j in in(i)} a[j] - indeg(i)*bb[i] + c3[i] )
__global__ void k_fitness(const float* __restrict__ A, const float* __restrict__ a,
                          const float* __restrict__ bbv, const float* __restrict__ c3,
                          float* __restrict__ fit, int N) {
  const int i = blockIdx.x, t = threadIdx.x;
  __shared__ float rs[256]; __shared__ int rc[256];
  float s = 0.f; int c = 0;
  for (int j = t; j < N; j += 256)
    if (A[(size_t)i * N + j] != 0.f || j == i) { s += a[j]; c++; }
  rs[t] = s; rc[t] = c; __syncthreads();
  for (int sh = 128; sh > 0; sh >>= 1) {
    if (t < sh) { rs[t] += rs[t + sh]; rc[t] += rc[t + sh]; }
    __syncthreads();
  }
  if (t == 0) {
    float agg = rs[0] - (float)rc[0] * bbv[i] + c3[i];
    fit[i] = 1.f / (1.f + __expf(-agg));
  }
}

// Per-graph top-k (descending, ties -> lowest index), O(n*k) selection.
__global__ void k_topk(const float* __restrict__ fit, int* __restrict__ perm,
                       int nper, int k) {
  const int g = blockIdx.x, t = threadIdx.x;
  __shared__ float vals[256];
  __shared__ float rv[256]; __shared__ int ri[256];
  vals[t] = (t < nper) ? fit[g * nper + t] : -1e30f;
  __syncthreads();
  for (int r = 0; r < k; ++r) {
    rv[t] = vals[t]; ri[t] = t; __syncthreads();
    for (int sh = 128; sh > 0; sh >>= 1) {
      if (t < sh) {
        if (rv[t + sh] > rv[t] || (rv[t + sh] == rv[t] && ri[t + sh] < ri[t])) {
          rv[t] = rv[t + sh]; ri[t] = ri[t + sh];
        }
      }
      __syncthreads();
    }
    if (t == 0) { perm[g * k + r] = g * nper + ri[0]; vals[ri[0]] = -1e30f; }
    __syncthreads();
  }
}

__global__ void k_gather_scale(const float* __restrict__ XN, const float* __restrict__ fit,
                               const int* __restrict__ perm, float* __restrict__ Xo, int d) {
  const int r = blockIdx.x, f = threadIdx.x;
  const int p = perm[r];
  Xo[(size_t)r * d + f] = XN[(size_t)p * d + f] * fit[p];
}

__global__ void k_gather_rows(const __bf16* __restrict__ St, const int* __restrict__ perm,
                              __bf16* __restrict__ Ss, int N) {
  const int r = blockIdx.y;
  const int j = blockIdx.x * 256 + threadIdx.x;
  if (j < N) Ss[(size_t)r * N + j] = St[(size_t)perm[r] * N + j];
}

__global__ void k_meanpool(const float* __restrict__ X, float* __restrict__ hg,
                           int nper, int d) {
  const int g = blockIdx.x, f = threadIdx.x;
  float s = 0.f;
  for (int r = 0; r < nper; ++r) s += X[(size_t)(g * nper + r) * d + f];
  hg[(size_t)g * d + f] = s / (float)nper;
}

// =====================================================================
// Host orchestration
// =====================================================================
extern "C" void kernel_launch(void* const* d_in, const int* in_sizes, int n_in,
                              void* d_out, int out_size, void* d_ws, size_t ws_size,
                              hipStream_t stream) {
  (void)n_in; (void)out_size; (void)ws_size;
  const float* x_in = (const float*)d_in[0];
  const int*   ei   = (const int*)d_in[1];
  const int    E    = in_sizes[1] / 2;
  auto P = [&](int i) { return (const float*)d_in[i]; };

  const int N0 = 4096, G = 16, DMAX = 256;

  // Param leaf indices (dict-insertion-order flattening).
  static const int gbase[4]   = {3, 24, 45, 66};
  static const int gdin[4][3] = {{39, 64, 64}, {128, 128, 128}, {128, 256, 256}, {256, 256, 256}};
  static const int gdout[4][3]= {{64, 64, 128}, {128, 128, 128}, {256, 256, 256}, {256, 256, 256}};
  static const int abase[3]   = {15, 36, 57};     // lin_w,lin_b,att_w,att_b,le1_w,le1_b,le2_w,le3_w,le3_b
  static const int adim[3]    = {128, 128, 256};
  const int FC = 78;                               // w1,b1,w2,b2

  // Workspace bump allocator (~200 MB peak).
  char* base = (char*)d_ws; size_t off = 0;
  auto alloc = [&](size_t bytes) -> void* {
    void* p = base + off; off += (bytes + 255) & ~(size_t)255; return p;
  };
  float*  A    = (float*) alloc((size_t)N0 * N0 * 4);        // dense binary adjacency
  __bf16* Ah   = (__bf16*)alloc((size_t)N0 * N0 * 2);        // Anorm / Asl (bf16, symmetric)
  __bf16* St   = (__bf16*)alloc((size_t)N0 * N0 * 2);        // softmax scores S^T (K-major rows)
  __bf16* Ssel = (__bf16*)alloc((size_t)(N0 / 2) * N0 * 2);  // selected rows of St
  __bf16* M1   = (__bf16*)alloc((size_t)(N0 / 2) * N0 * 2);  // Ssel @ Asl (bf16)
  float*  Anew = (float*) alloc((size_t)(N0 / 2) * (N0 / 2) * 4);
  float*  di   = (float*) alloc((size_t)N0 * 4);
  float*  X    = (float*) alloc((size_t)N0 * DMAX * 4);      // current features
  float*  XN   = (float*) alloc((size_t)N0 * DMAX * 4);      // xnew
  float*  T    = (float*) alloc((size_t)N0 * DMAX * 4);      // GEMM temp
  float*  XQ   = (float*) alloc((size_t)N0 * DMAX * 4);      // segment-max
  __bf16* Xh   = (__bf16*)alloc((size_t)N0 * DMAX * 2);      // bf16 operand (A side / X^T)
  __bf16* Th   = (__bf16*)alloc((size_t)N0 * DMAX * 2);      // bf16 operand (T^T etc.)
  __bf16* Qh   = (__bf16*)alloc((size_t)N0 * DMAX * 2);      // bf16 xq
  __bf16* Wh   = (__bf16*)alloc((size_t)65536 * 2);          // transposed weight
  float*  uvec = (float*) alloc((size_t)N0 * 4);
  float*  wvec = (float*) alloc((size_t)N0 * 4);
  float*  avec = (float*) alloc((size_t)N0 * 4);
  float*  bvec = (float*) alloc((size_t)N0 * 4);
  float*  cvec = (float*) alloc((size_t)N0 * 4);
  float*  fit  = (float*) alloc((size_t)N0 * 4);
  int*    perm = (int*)   alloc((size_t)(N0 / 2) * 4);
  float*  hg   = (float*) alloc((size_t)G * 256 * 4);
  float*  t1   = (float*) alloc((size_t)G * 256 * 4);

  // C = A(MxKp) * Bt(N x Kp)^T, one wave per 32x32 tile, 4 tiles/block.
  auto gemm = [&](const __bf16* Aop, const __bf16* Bop, const float* bias,
                  float* Cf, __bf16* Ch, int M, int N, int Kp,
                  int lda, int ldb, int ldc, int flags) {
    dim3 grid((N + 31) / 32, ((M + 31) / 32 + 3) / 4);
    k_gemm<<<grid, dim3(32, 4), 0, stream>>>(Aop, Bop, bias, Cf, Ch, M, N, Kp, lda, ldb, ldc, flags);
  };
  // row-padded bf16 copy: src (R x C) -> dst (R x ldd)
  auto conv = [&](const float* s, __bf16* d, int R, int C, int ldd) {
    k_conv_pad<<<dim3((ldd + 255) / 256, R), dim3(256), 0, stream>>>(s, d, C, ldd);
  };
  // transposed bf16 copy: src (R x C) -> dst (Cp x ldd), zero padded
  auto convT = [&](const float* s, __bf16* d, int R, int C, int ldd, int Cp) {
    k_convT<<<dim3((ldd + 255) / 256, Cp), dim3(256), 0, stream>>>(s, d, R, C, ldd);
  };

  // ---- build A0 from edge_index ----
  int N = N0;
  k_zero_f32<<<dim3((N * N + 255) / 256), dim3(256), 0, stream>>>(A, N * N);
  k_build_A0<<<dim3((E + 255) / 256), dim3(256), 0, stream>>>(ei, E, A, N);

  for (int blk = 0; blk < 4; ++blk) {
    // ---- GCN normalization: Ah = D^-1/2 (A+I) D^-1/2, bf16 ----
    k_degree<<<dim3(N), dim3(256), 0, stream>>>(A, di, N);
    k_anorm<<<dim3((N * N + 255) / 256), dim3(256), 0, stream>>>(A, di, Ah, N);

    // ---- 3 GCN layers: X = BN(relu(Ah @ (X@W) + b)) ----
    for (int L = 0; L < 3; ++L) {
      const int din = gdin[blk][L], dout = gdout[blk][L];
      const int dinp = (din + 31) & ~31;                 // K padded (39 -> 64)
      const int pi = gbase[blk] + 4 * L;
      convT(P(pi), Wh, din, dout, dinp, dout);           // W^T: dout x dinp
      conv((blk == 0 && L == 0) ? x_in : X, Xh, N, din, dinp);
      gemm(Xh, Wh, nullptr, T, nullptr, N, dout, dinp, dinp, dinp, dout, 0);   // T = X W
      convT(T, Th, N, dout, N, dout);                    // T^T: dout x N
      gemm(Ah, Th, P(pi + 1), T, nullptr, N, dout, N, N, N, dout, 1);          // Z = Ah T + b
      k_bn<<<dim3(dout), dim3(256), 0, stream>>>(T, P(pi + 2), P(pi + 3), X, N, dout);
    }
    if (blk == 3) break;

    // ---- ASAP pooling ----
    const int d = adim[blk], ab = abase[blk], K = N / 2, nper = N / G;
    k_segmax<<<dim3(N), dim3(d), 0, stream>>>(A, X, XQ, N, d);
    conv(XQ, Qh, N, d, d);
    convT(P(ab + 0), Wh, d, d, d, d);
    gemm(Qh, Wh, P(ab + 1), T, nullptr, N, d, d, d, d, d, 1);                  // xq_e
    k_rowdot<<<dim3(N), dim3(256), 0, stream>>>(T, P(ab + 2),     P(ab + 3), uvec, d);
    k_rowdot<<<dim3(N), dim3(256), 0, stream>>>(X, P(ab + 2) + d, nullptr,   wvec, d);
    k_scores<<<dim3(N), dim3(256), 0, stream>>>(A, uvec, wvec, St, N);
    convT(X, Xh, N, d, N, d);                                                  // X^T: d x N
    gemm(St, Xh, nullptr, XN, nullptr, N, d, N, N, N, d, 0);                   // xnew = S^T X
    k_rowdot<<<dim3(N), dim3(256), 0, stream>>>(XN, P(ab + 4), P(ab + 5), avec, d); // le1
    k_rowdot<<<dim3(N), dim3(256), 0, stream>>>(XN, P(ab + 6), nullptr,   bvec, d); // le2
    k_rowdot<<<dim3(N), dim3(256), 0, stream>>>(XN, P(ab + 7), P(ab + 8), cvec, d); // le3
    k_fitness<<<dim3(N), dim3(256), 0, stream>>>(A, avec, bvec, cvec, fit, N);
    k_topk<<<dim3(G), dim3(256), 0, stream>>>(fit, perm, nper, nper / 2);
    k_gather_scale<<<dim3(K), dim3(d), 0, stream>>>(XN, fit, perm, X, d);
    k_gather_rows<<<dim3((N + 255) / 256, K), dim3(256), 0, stream>>>(St, perm, Ssel, N);
    // ---- coarsen: A' = binarize( Ssel (A+I) Ssel^T ), offdiag ----
    k_asl<<<dim3((N * N + 255) / 256), dim3(256), 0, stream>>>(A, Ah, N);
    gemm(Ssel, Ah, nullptr, nullptr, M1, K, N, N, N, N, N, 8);   // (A+I) symmetric => B^T = B
    gemm(M1, Ssel, nullptr, Anew, nullptr, K, K, N, N, N, K, 0); // M1 Ssel^T (K-major B natively)
    k_binarize<<<dim3((K * K + 255) / 256), dim3(256), 0, stream>>>(Anew, A, K);
    N = K;
  }

  // ---- global mean pool + FC head ----
  k_meanpool<<<dim3(G), dim3(256), 0, stream>>>(X, hg, N / G, 256);
  conv(hg, Xh, G, 256, 256);
  convT(P(FC + 0), Wh, 256, 256, 256, 256);
  gemm(Xh, Wh, P(FC + 1), t1, nullptr, G, 256, 256, 256, 256, 256, 1 | 2);
  conv(t1, Th, G, 256, 256);
  convT(P(FC + 2), Wh, 256, 40, 256, 48);                 // pad 40 -> 48 rows
  gemm(Th, Wh, P(FC + 3), (float*)d_out, nullptr, G, 40, 256, 256, 256, 40, 1);
}
